// GraphTransformerNet_10977936408581
// MI455X (gfx1250) — compile-verified, matrix operand order checked by reading, not run on hardware
//
#include <hip/hip_runtime.h>
#include <cstdint>
#include <cstddef>

// ---------------------------------------------------------------------------
// Problem constants (from reference)
// ---------------------------------------------------------------------------
#define NNODES 100000
#define NEDGES 300000
#define NGRAPH 4096
#define FN_    64
#define FE_    16
#define FP_    1034
#define HDIM   256
#define NHEAD  4
#define CHAN   64
#define KCAT   1344   // 256+1034 = 1290 padded up to 42*32

typedef __attribute__((ext_vector_type(16))) __bf16 v16bf;
typedef __attribute__((ext_vector_type(8)))  float  v8f;

union BfOp { uint4 q[2]; v16bf v; };

__device__ __forceinline__ unsigned short f2bf(float f) {
  unsigned u = __float_as_uint(f);
  u += 0x7FFFu + ((u >> 16) & 1u);           // round-to-nearest-even
  return (unsigned short)(u >> 16);
}
__device__ __forceinline__ unsigned encf(float f) {   // order-preserving map
  unsigned u = __float_as_uint(f);
  return (u & 0x80000000u) ? ~u : (u | 0x80000000u);
}
__device__ __forceinline__ float decf(unsigned u) {
  return __uint_as_float((u & 0x80000000u) ? (u & 0x7FFFFFFFu) : ~u);
}

// ---------------------------------------------------------------------------
// CDNA5 async memory->LDS staging of We (16 KB = FE_*HDIM*4B).
// 256 threads x 16B x 4 slices; IOFFSET applies to BOTH global and LDS addr
// (ISA 10.x async pseudocode), so one base VGPR pair + 4 offsets covers it.
// Tracked by ASYNCcnt -> s_wait_asynccnt 0, then workgroup barrier.
// ---------------------------------------------------------------------------
__device__ __forceinline__ void stage_we_async(const float* __restrict__ We,
                                               float* sWe)
{
  unsigned long long ga = (unsigned long long)(const void*)We
                        + (unsigned long long)threadIdx.x * 16ull;
  unsigned lds = (unsigned)(uintptr_t)(void*)sWe + (unsigned)threadIdx.x * 16u;
  asm volatile("global_load_async_to_lds_b128 %0, %1, off"
               :: "v"(lds), "v"(ga) : "memory");
  asm volatile("global_load_async_to_lds_b128 %0, %1, off offset:4096"
               :: "v"(lds), "v"(ga) : "memory");
  asm volatile("global_load_async_to_lds_b128 %0, %1, off offset:8192"
               :: "v"(lds), "v"(ga) : "memory");
  asm volatile("global_load_async_to_lds_b128 %0, %1, off offset:12288"
               :: "v"(lds), "v"(ga) : "memory");
  asm volatile("s_wait_asynccnt 0x0" ::: "memory");
  __syncthreads();
}

// ---------------------------------------------------------------------------
// Generic WMMA GEMM:  C[M,Ncols] = A_bf16[M,K] * W  (+bias), W packed as
// Wp[col][k] (bf16, transposed) so B-operand columns load contiguously.
// Block = 8 waves -> 32-row x 256-col tile; wave = 16x64 strip (4 wmma accs).
// K multiple of 32; M multiple of 32 (100000 = 3125*32, 4096 = 128*32).
// ---------------------------------------------------------------------------
__global__ __launch_bounds__(256) void wmma_gemm(
    const unsigned short* __restrict__ A,
    const unsigned short* __restrict__ Bp,
    const float* __restrict__ bias,
    float* __restrict__ C,
    int M, int K, int Ncols)
{
  const int lane = threadIdx.x & 31;
  const int wave = threadIdx.x >> 5;
  const int l16  = lane & 15;
  const int hi   = lane >> 4;                     // 0: K 0-7/16-23, 1: K 8-15/24-31
  const int rowBase = blockIdx.x * 32 + (wave & 1) * 16;
  const int colBase = blockIdx.y * 256 + (wave >> 1) * 64;
  if (rowBase + 16 > M) return;                   // wave-uniform, never splits EXEC

  const unsigned short* aP  = A  + (size_t)(rowBase + l16) * K + hi * 8;
  const unsigned short* bP0 = Bp + (size_t)(colBase +  0 + l16) * K + hi * 8;
  const unsigned short* bP1 = Bp + (size_t)(colBase + 16 + l16) * K + hi * 8;
  const unsigned short* bP2 = Bp + (size_t)(colBase + 32 + l16) * K + hi * 8;
  const unsigned short* bP3 = Bp + (size_t)(colBase + 48 + l16) * K + hi * 8;

  v8f acc0 = {}, acc1 = {}, acc2 = {}, acc3 = {};
  for (int kb = 0; kb < K; kb += 32) {
    __builtin_prefetch(aP + kb + 256, 0, 0);      // global_prefetch_b8 (A stream)
    BfOp a;  a.q[0]  = *(const uint4*)(aP  + kb); a.q[1]  = *(const uint4*)(aP  + kb + 16);
    BfOp b0; b0.q[0] = *(const uint4*)(bP0 + kb); b0.q[1] = *(const uint4*)(bP0 + kb + 16);
    BfOp b1; b1.q[0] = *(const uint4*)(bP1 + kb); b1.q[1] = *(const uint4*)(bP1 + kb + 16);
    BfOp b2; b2.q[0] = *(const uint4*)(bP2 + kb); b2.q[1] = *(const uint4*)(bP2 + kb + 16);
    BfOp b3; b3.q[0] = *(const uint4*)(bP3 + kb); b3.q[1] = *(const uint4*)(bP3 + kb + 16);
    acc0 = __builtin_amdgcn_wmma_f32_16x16x32_bf16(false, a.v, false, b0.v, (short)0, acc0, false, false);
    acc1 = __builtin_amdgcn_wmma_f32_16x16x32_bf16(false, a.v, false, b1.v, (short)0, acc1, false, false);
    acc2 = __builtin_amdgcn_wmma_f32_16x16x32_bf16(false, a.v, false, b2.v, (short)0, acc2, false, false);
    acc3 = __builtin_amdgcn_wmma_f32_16x16x32_bf16(false, a.v, false, b3.v, (short)0, acc3, false, false);
  }

  #pragma unroll
  for (int j = 0; j < 4; ++j) {
    v8f accv = (j == 0) ? acc0 : (j == 1) ? acc1 : (j == 2) ? acc2 : acc3;
    const int col = colBase + j * 16 + l16;
    const float bv = bias[col];
    float* cP = C + (size_t)(rowBase + hi * 8) * Ncols + col;
    #pragma unroll
    for (int i = 0; i < 8; ++i)                   // VGPR i: M = i (+8 for hi lanes)
      cP[(size_t)i * Ncols] = accv[i] + bv;
  }
}

// ---------------------------------------------------------------------------
// Weight packing: fused [Wq|Wk|Wv|Ws] (each [K,256] f32) -> Wp[1024][K] bf16
// ---------------------------------------------------------------------------
__global__ void pack_qkvs(const float* __restrict__ Wq, const float* __restrict__ bq,
                          const float* __restrict__ Wk, const float* __restrict__ bk,
                          const float* __restrict__ Wv, const float* __restrict__ bv,
                          const float* __restrict__ Ws, const float* __restrict__ bs,
                          unsigned short* __restrict__ Wp, float* __restrict__ biasAll,
                          int K)
{
  const int col = blockIdx.x;                     // 0..1023
  const int seg = col >> 8, c = col & 255;
  const float* W = (seg == 0) ? Wq : (seg == 1) ? Wk : (seg == 2) ? Wv : Ws;
  const float* b = (seg == 0) ? bq : (seg == 1) ? bk : (seg == 2) ? bv : bs;
  for (int k = threadIdx.x; k < K; k += blockDim.x)
    Wp[(size_t)col * K + k] = f2bf(W[(size_t)k * 256 + c]);
  if (threadIdx.x == 0) biasAll[col] = b[c];
}

__global__ void pack_w1(const float* __restrict__ W1, unsigned short* __restrict__ Wp)
{
  const int col = blockIdx.x;                     // 0..255
  for (int k = threadIdx.x; k < KCAT; k += blockDim.x)
    Wp[(size_t)col * KCAT + k] = (k < 1290) ? f2bf(W1[(size_t)k * 256 + col])
                                            : (unsigned short)0;
}

// ---------------------------------------------------------------------------
// Elementwise helpers
// ---------------------------------------------------------------------------
__global__ void cvt_bf16_k(const float* __restrict__ in, unsigned short* __restrict__ out,
                           size_t n)
{
  size_t i = (size_t)blockIdx.x * blockDim.x + threadIdx.x;
  if (i < n) out[i] = f2bf(in[i]);
}

// h_next = relu(msg_acc + skip)  (skip = qkvs col block 768..1023); dual write f32+bf16
__global__ void relu_cvt(const float* __restrict__ msgacc, const float* __restrict__ qkvs,
                         float* __restrict__ hf, unsigned short* __restrict__ hb)
{
  size_t i = (size_t)blockIdx.x * blockDim.x + threadIdx.x;
  if (i >= (size_t)NNODES * HDIM) return;
  size_t n = i >> 8; int c = (int)(i & 255);
  float v = msgacc[i] + qkvs[n * 1024 + 768 + c];
  v = fmaxf(v, 0.f);
  hf[i] = v; hb[i] = f2bf(v);
}

__global__ void init_attn(unsigned* __restrict__ amax, float* __restrict__ denom,
                          float* __restrict__ msgacc)
{
  size_t i = (size_t)blockIdx.x * blockDim.x + threadIdx.x;
  if (i < (size_t)NNODES * NHEAD) { amax[i] = encf(-3.0e38f); denom[i] = 0.f; }
  if (i < (size_t)NNODES * HDIM)  msgacc[i] = 0.f;
}

// ---------------------------------------------------------------------------
// Attention pass 1: alpha_raw[e,h] = q[dst].(k[src]+We(ea)) / 8 ; atomicMax amax
// One wave per edge; lane = head*8 + chunk, 8 channels each. We staged in LDS
// via CDNA5 async global->LDS copy.
// ---------------------------------------------------------------------------
__global__ __launch_bounds__(256) void edge_alpha(
    const int* __restrict__ srcI, const int* __restrict__ dstI,
    const float* __restrict__ ea, const float* __restrict__ We,
    const float* __restrict__ qkvs, float* __restrict__ alpha_raw,
    unsigned* __restrict__ amax, int E)
{
  __shared__ float sWe[FE_ * HDIM];
  stage_we_async(We, sWe);

  const int e = blockIdx.x * 8 + (threadIdx.x >> 5);
  if (e >= E) return;
  const int lane = threadIdx.x & 31;
  const int h = lane >> 3, ci = lane & 7;
  const int s = srcI[e], d = dstI[e];
  const int cb = h * CHAN + ci * 8;

  float eaL[FE_];
  #pragma unroll
  for (int j = 0; j < FE_; ++j) eaL[j] = ea[(size_t)e * FE_ + j];

  float part = 0.f;
  #pragma unroll
  for (int t = 0; t < 8; ++t) {
    const int c = cb + t;
    float ep = 0.f;
    #pragma unroll
    for (int j = 0; j < FE_; ++j) ep = fmaf(eaL[j], sWe[j * HDIM + c], ep);
    const float kv = qkvs[(size_t)s * 1024 + 256 + c] + ep;
    part = fmaf(qkvs[(size_t)d * 1024 + c], kv, part);
  }
  part += __shfl_xor(part, 1, 32);
  part += __shfl_xor(part, 2, 32);
  part += __shfl_xor(part, 4, 32);
  if (ci == 0) {
    const float a = part * 0.125f;                // / sqrt(64)
    alpha_raw[(size_t)e * NHEAD + h] = a;
    atomicMax(&amax[(size_t)d * NHEAD + h], encf(a));
  }
}

// Pass 2: ex = exp(alpha - amax[dst]); denom += ex
__global__ void edge_exp(const int* __restrict__ dstI, const float* __restrict__ alpha_raw,
                         const unsigned* __restrict__ amax, float* __restrict__ exbuf,
                         float* __restrict__ denom, int E)
{
  const int i = blockIdx.x * blockDim.x + threadIdx.x;
  if (i >= E * NHEAD) return;
  const int e = i >> 2, h = i & 3;
  const int d = dstI[e];
  const float ex = __expf(alpha_raw[i] - decf(amax[(size_t)d * NHEAD + h]));
  exbuf[i] = ex;
  atomicAdd(&denom[(size_t)d * NHEAD + h], ex);
}

// Pass 3: msg = (v[src]+We(ea)) * ex/denom[dst]; scatter-add into msgacc[dst]
__global__ __launch_bounds__(256) void edge_msg(
    const int* __restrict__ srcI, const int* __restrict__ dstI,
    const float* __restrict__ ea, const float* __restrict__ We,
    const float* __restrict__ qkvs, const float* __restrict__ exbuf,
    const float* __restrict__ denom, float* __restrict__ msgacc, int E)
{
  __shared__ float sWe[FE_ * HDIM];
  stage_we_async(We, sWe);

  const int e = blockIdx.x * 8 + (threadIdx.x >> 5);
  if (e >= E) return;
  const int lane = threadIdx.x & 31;
  const int h = lane >> 3, ci = lane & 7;
  const int s = srcI[e], d = dstI[e];
  const int cb = h * CHAN + ci * 8;
  const float coef = exbuf[(size_t)e * NHEAD + h] /
                     fmaxf(denom[(size_t)d * NHEAD + h], 1e-30f);

  float eaL[FE_];
  #pragma unroll
  for (int j = 0; j < FE_; ++j) eaL[j] = ea[(size_t)e * FE_ + j];

  #pragma unroll
  for (int t = 0; t < 8; ++t) {
    const int c = cb + t;
    float ep = 0.f;
    #pragma unroll
    for (int j = 0; j < FE_; ++j) ep = fmaf(eaL[j], sWe[j * HDIM + c], ep);
    const float msg = (qkvs[(size_t)s * 1024 + 512 + c] + ep) * coef;
    atomicAdd(&msgacc[(size_t)d * HDIM + c], msg);
  }
}

// ---------------------------------------------------------------------------
// Pooling + head
// ---------------------------------------------------------------------------
__global__ void pool_init(float* __restrict__ pooled, float* __restrict__ cnt)
{
  size_t i = (size_t)blockIdx.x * blockDim.x + threadIdx.x;
  if (i < (size_t)NGRAPH * HDIM) pooled[i] = 0.f;
  if (i < NGRAPH) cnt[i] = 0.f;
}

__global__ void pool_sum(const float* __restrict__ h, const int* __restrict__ batch,
                         float* __restrict__ pooled, float* __restrict__ cnt)
{
  size_t i = (size_t)blockIdx.x * blockDim.x + threadIdx.x;
  if (i >= (size_t)NNODES * HDIM) return;
  const int n = (int)(i >> 8), c = (int)(i & 255);
  const int b = batch[n];
  atomicAdd(&pooled[(size_t)b * HDIM + c], h[i]);
  if (c == 0) atomicAdd(&cnt[b], 1.0f);
}

__global__ void build_zcat(const float* __restrict__ pooled, const float* __restrict__ cnt,
                           const float* __restrict__ fp, unsigned short* __restrict__ zcat)
{
  const int g = blockIdx.x;
  const float inv = 1.0f / fmaxf(cnt[g], 1.0f);
  for (int c = threadIdx.x; c < KCAT; c += blockDim.x) {
    float v;
    if (c < HDIM)        v = pooled[(size_t)g * HDIM + c] * inv;
    else if (c < 1290)   v = fp[(size_t)g * FP_ + (c - HDIM)];
    else                 v = 0.f;
    zcat[(size_t)g * KCAT + c] = f2bf(v);
  }
}

__global__ void bn_stats(const float* __restrict__ z, float* __restrict__ mu,
                         float* __restrict__ var)
{
  __shared__ float sh[256], sh2[256];
  const int c = blockIdx.x, t = threadIdx.x;
  float s = 0.f, s2 = 0.f;
  for (int g = t; g < NGRAPH; g += 256) {
    const float v = z[(size_t)g * HDIM + c];
    s += v; s2 += v * v;
  }
  sh[t] = s; sh2[t] = s2; __syncthreads();
  for (int off = 128; off; off >>= 1) {
    if (t < off) { sh[t] += sh[t + off]; sh2[t] += sh2[t + off]; }
    __syncthreads();
  }
  if (t == 0) {
    const float m = sh[0] * (1.0f / NGRAPH);
    mu[c] = m;
    var[c] = sh2[0] * (1.0f / NGRAPH) - m * m;
  }
}

__global__ void head_out(const float* __restrict__ z, const float* __restrict__ mu,
                         const float* __restrict__ var, const float* __restrict__ gamma,
                         const float* __restrict__ beta, const float* __restrict__ W2,
                         const float* __restrict__ b2, float* __restrict__ out)
{
  const int g = blockIdx.x * blockDim.x + threadIdx.x;
  if (g >= NGRAPH) return;
  float acc = b2[0];
  for (int c = 0; c < HDIM; ++c) {
    const float zn = (z[(size_t)g * HDIM + c] - mu[c]) * rsqrtf(var[c] + 1e-5f) * gamma[c]
                     + beta[c];
    acc = fmaf(fmaxf(zn, 0.f), W2[c], acc);
  }
  out[g] = acc;
}

// ---------------------------------------------------------------------------
// Host orchestration
// ---------------------------------------------------------------------------
extern "C" void kernel_launch(void* const* d_in, const int* in_sizes, int n_in,
                              void* d_out, int out_size, void* d_ws, size_t ws_size,
                              hipStream_t stream)
{
  (void)in_sizes; (void)n_in; (void)out_size; (void)ws_size;
  constexpr int N = NNODES, E = NEDGES, G = NGRAPH;

  // ---- inputs (setup_inputs dict insertion order) ----
  const float* x      = (const float*)d_in[0];
  const int*   eidx   = (const int*)d_in[1];
  const int*   srcI   = eidx;
  const int*   dstI   = eidx + E;
  const float* eattr  = (const float*)d_in[2];
  const float* fprint = (const float*)d_in[3];
  const int*   batch  = (const int*)d_in[4];
  // layers: 9 tensors each starting at index 5
  const float* W1    = (const float*)d_in[5 + 4 * 9 + 0];
  const float* b1    = (const float*)d_in[5 + 4 * 9 + 1];
  const float* gamma = (const float*)d_in[5 + 4 * 9 + 2];
  const float* beta  = (const float*)d_in[5 + 4 * 9 + 3];
  const float* W2    = (const float*)d_in[5 + 4 * 9 + 4];
  const float* b2    = (const float*)d_in[5 + 4 * 9 + 5];
  float* out = (float*)d_out;

  // ---- workspace carve-up ----
  char* p = (char*)d_ws;
  auto alloc = [&](size_t bytes) -> void* {
    void* r = (void*)p;
    p += (bytes + 255) & ~(size_t)255;
    return r;
  };
  unsigned short* hb     = (unsigned short*)alloc((size_t)N * HDIM * 2); // bf16 activations
  float*          qkvs   = (float*)alloc((size_t)N * 1024 * 4);          // Q|K|V|S
  float*          msgacc = (float*)alloc((size_t)N * HDIM * 4);
  float*          hf     = (float*)alloc((size_t)N * HDIM * 4);
  float*          araw   = (float*)alloc((size_t)E * NHEAD * 4);
  float*          exbuf  = (float*)alloc((size_t)E * NHEAD * 4);
  unsigned*       amax   = (unsigned*)alloc((size_t)N * NHEAD * 4);
  float*          denom  = (float*)alloc((size_t)N * NHEAD * 4);
  unsigned short* Wp     = (unsigned short*)alloc((size_t)1024 * HDIM * 2);
  float*          biasA  = (float*)alloc(1024 * 4);
  float*          pooled = (float*)alloc((size_t)G * HDIM * 4);
  float*          cnt    = (float*)alloc((size_t)G * 4);
  unsigned short* zcat   = (unsigned short*)alloc((size_t)G * KCAT * 2);
  unsigned short* Wp1    = (unsigned short*)alloc((size_t)HDIM * KCAT * 2);
  float*          z      = (float*)alloc((size_t)G * HDIM * 4);
  float*          mu     = (float*)alloc(HDIM * 4);
  float*          var    = (float*)alloc(HDIM * 4);

  const int BLK = 256;
  auto gs = [](size_t n, int b) { return (unsigned)((n + (size_t)b - 1) / b); };

  // layer 0 A-matrix: x -> bf16
  cvt_bf16_k<<<gs((size_t)N * FN_, BLK), BLK, 0, stream>>>(x, hb, (size_t)N * FN_);

  for (int l = 0; l < 4; ++l) {
    const int base = 5 + l * 9;
    const float* Wq = (const float*)d_in[base + 0];
    const float* bq = (const float*)d_in[base + 1];
    const float* Wk = (const float*)d_in[base + 2];
    const float* bk = (const float*)d_in[base + 3];
    const float* Wv = (const float*)d_in[base + 4];
    const float* bv = (const float*)d_in[base + 5];
    const float* We = (const float*)d_in[base + 6];
    const float* Ws = (const float*)d_in[base + 7];
    const float* bs = (const float*)d_in[base + 8];
    const int K = (l == 0) ? FN_ : HDIM;

    pack_qkvs<<<1024, BLK, 0, stream>>>(Wq, bq, Wk, bk, Wv, bv, Ws, bs, Wp, biasA, K);
    wmma_gemm<<<dim3(N / 32, 4), BLK, 0, stream>>>(hb, Wp, biasA, qkvs, N, K, 1024);

    init_attn<<<gs((size_t)N * HDIM, BLK), BLK, 0, stream>>>(amax, denom, msgacc);
    edge_alpha<<<gs(E, 8), BLK, 0, stream>>>(srcI, dstI, eattr, We, qkvs, araw, amax, E);
    edge_exp<<<gs((size_t)E * NHEAD, BLK), BLK, 0, stream>>>(dstI, araw, amax, exbuf, denom, E);
    edge_msg<<<gs(E, 8), BLK, 0, stream>>>(srcI, dstI, eattr, We, qkvs, exbuf, denom, msgacc, E);

    relu_cvt<<<gs((size_t)N * HDIM, BLK), BLK, 0, stream>>>(msgacc, qkvs, hf, hb);
  }

  // global mean pool
  pool_init<<<gs((size_t)G * HDIM, BLK), BLK, 0, stream>>>(pooled, cnt);
  pool_sum<<<gs((size_t)N * HDIM, BLK), BLK, 0, stream>>>(hf, batch, pooled, cnt);
  build_zcat<<<G, BLK, 0, stream>>>(pooled, cnt, fprint, zcat);

  // head GEMM [G,1344] x [1344,256]
  pack_w1<<<HDIM, BLK, 0, stream>>>(W1, Wp1);
  wmma_gemm<<<dim3(G / 32, 1), BLK, 0, stream>>>(zcat, Wp1, b1, z, G, KCAT, HDIM);

  // batchnorm stats + final projection
  bn_stats<<<HDIM, 256, 0, stream>>>(z, mu, var);
  head_out<<<gs(G, BLK), BLK, 0, stream>>>(z, mu, var, gamma, beta, W2, b2, out);
}
